// NCEAverageKineticsMining_67912022885097
// MI455X (gfx1250) — matrix operand order repaired and checked by reading, not compiled, not run on hardware
//
#include <hip/hip_runtime.h>
#include <hip/hip_bf16.h>
#include <stdint.h>

// CDNA5 wave32 WMMA types (probe-confirmed signatures)
typedef __attribute__((ext_vector_type(16))) __bf16 v16bf;
typedef __attribute__((ext_vector_type(8)))  float  v8f;

#define B_      256
#define N_      262144
#define D_      128
#define K_      4096
#define TOPE    (B_*(K_+1))
#define INV_T   (1.0f/0.07f)
#define TIE_MAX 4096

// Monotonic float -> uint mapping (descending top-k == largest keys)
__device__ __forceinline__ unsigned f32_key(float f) {
  unsigned u = __float_as_uint(f);
  return (u & 0x80000000u) ? ~u : (u | 0x80000000u);
}

// A-fragment (16x32 bf16, contiguous k order) from a row-major fp32 row pointer.
// Per lane: e=0..15 -> d = 32*ks + 16*lhalf + e
// 16 contiguous floats -> 4x float4 loads + packed converts.
__device__ __forceinline__ v16bf load_a_frag(const float* __restrict__ rowp,
                                             int ks, int lhalf) {
  const float4* p0 = (const float4*)(rowp + 32 * ks + 16 * lhalf);
  const float4* p1 = (const float4*)(rowp + 32 * ks + 16 * lhalf + 8);
  float4 a0 = p0[0], a1 = p0[1], b0 = p1[0], b1 = p1[1];
  v16bf v;
  v[0]  = (__bf16)a0.x; v[1]  = (__bf16)a0.y; v[2]  = (__bf16)a0.z; v[3]  = (__bf16)a0.w;
  v[4]  = (__bf16)a1.x; v[5]  = (__bf16)a1.y; v[6]  = (__bf16)a1.z; v[7]  = (__bf16)a1.w;
  v[8]  = (__bf16)b0.x; v[9]  = (__bf16)b0.y; v[10] = (__bf16)b0.z; v[11] = (__bf16)b0.w;
  v[12] = (__bf16)b1.x; v[13] = (__bf16)b1.y; v[14] = (__bf16)b1.z; v[15] = (__bf16)b1.w;
  return v;
}

// B-fragment (32x16 bf16) where every column is the same fp32 vector (rowp):
// K = 32*ks + 16*lhalf + e -> 16 contiguous floats -> 4x float4 loads.
__device__ __forceinline__ v16bf load_b_frag_bcast(const float* __restrict__ rowp,
                                                   int ks, int lhalf) {
  const float4* p = (const float4*)(rowp + 32 * ks + 16 * lhalf);
  float4 x0 = p[0], x1 = p[1], x2 = p[2], x3 = p[3];
  v16bf v;
  v[0]  = (__bf16)x0.x; v[1]  = (__bf16)x0.y; v[2]  = (__bf16)x0.z; v[3]  = (__bf16)x0.w;
  v[4]  = (__bf16)x1.x; v[5]  = (__bf16)x1.y; v[6]  = (__bf16)x1.z; v[7]  = (__bf16)x1.w;
  v[8]  = (__bf16)x2.x; v[9]  = (__bf16)x2.y; v[10] = (__bf16)x2.z; v[11] = (__bf16)x2.w;
  v[12] = (__bf16)x3.x; v[13] = (__bf16)x3.y; v[14] = (__bf16)x3.z; v[15] = (__bf16)x3.w;
  return v;
}

// ---------------------------------------------------------------------------
// Phase 1: scores[b,n] = sum_d anchor[b,d] * mem[d*N + n]   (memR reshape!)
// Block = 512 thr = 16 waves; wave w owns b-tile w and keeps its A fragments
// in registers for the whole block. The 128x16 fp32 bank tile is staged into
// LDS *already permuted into B-fragment order* [ks][lane][e] (bf16), so each
// wave pulls a whole v16bf fragment with one vector LDS load.
__global__ __launch_bounds__(512)
void gemm_scores_kernel(const float* __restrict__ anchor,
                        const float* __restrict__ mem,
                        const float* __restrict__ thr,
                        unsigned* __restrict__ scoresU) {
  __shared__ __bf16 ldsFrag[4 * 32 * 16];   // [ks][lane][e], 4 KB
  const int tid   = threadIdx.x;
  const int wave  = tid >> 5;      // b-tile 0..15
  const int lane  = tid & 31;
  const int lrow  = lane & 15;
  const int lhalf = lane >> 4;

  // A fragments held for the whole block
  v16bf Av[4];
  {
    const float* arow = anchor + (wave * 16 + lrow) * D_;
#pragma unroll
    for (int ks = 0; ks < 4; ++ks) Av[ks] = load_a_frag(arow, ks, lhalf);
  }
  const float thv = thr[0];

  // Staging geometry: thread handles column `col`, d-range [4*dgrp, 4*dgrp+3].
  const int col  = tid & 15;
  const int dgrp = tid >> 4;       // 0..31
  const int d0   = dgrp * 4;
  // destination inside fragment-ordered LDS (4 consecutive e slots, 8B store)
  const int ksS  = d0 >> 5;
  const int hS   = (d0 >> 4) & 1;
  const int posS = ksS * 512 + hS * 256 + col * 16 + (d0 & 15);

  const v16bf* fragp = (const v16bf*)ldsFrag;

  for (int nt = 0; nt < 16; ++nt) {
    const int nbase = (blockIdx.x * 16 + nt) * 16;
    __syncthreads();   // previous iteration's LDS reads done
    {
      const float* src = mem + (size_t)d0 * N_ + nbase + col;
      __builtin_prefetch(src + 16, 0, 0);   // next n-tile -> global_prefetch_b8
      union { __bf16 h[4]; uint2 u; } pk;
#pragma unroll
      for (int q = 0; q < 4; ++q) pk.h[q] = (__bf16)src[(size_t)q * N_];
      *(uint2*)&ldsFrag[posS] = pk.u;       // single 8B LDS store
    }
    __syncthreads();

    v8f c = {};
#pragma unroll
    for (int ks = 0; ks < 4; ++ks) {
      v16bf Bv = fragp[ks * 32 + lane];     // 32B vector LDS load
      c = __builtin_amdgcn_wmma_f32_16x16x32_bf16(false, Av[ks], false, Bv,
                                                  (short)0, c, false, false);
    }
    // C layout: c[j] = (M = j + 8*lhalf, N = lrow)
#pragma unroll
    for (int j = 0; j < 8; ++j) {
      float s = c[j];
      if (s >= thv) s = -2.0f;               // reference masking (dead at thr=10)
      int row = wave * 16 + j + 8 * lhalf;
      scoresU[(size_t)row * N_ + nbase + lrow] = f32_key(s);
    }
  }
}

// ---------------------------------------------------------------------------
// Phase 2a: exact radix select, one block per row, 3 passes (11/11/10 bits).
__global__ __launch_bounds__(1024)
void radix_pass_kernel(const unsigned* __restrict__ scoresU,
                       uint4* __restrict__ rowState, int pass) {
  __shared__ unsigned hist[2048];
  __shared__ unsigned sPref, sKrem;
  const int b   = blockIdx.x;
  const int tid = threadIdx.x;
  if (tid == 0) {
    if (pass == 0) { sPref = 0u; sKrem = K_; }
    else { uint4 st = rowState[b]; sPref = st.x; sKrem = st.y; }
  }
  for (int i = tid; i < 2048; i += 1024) hist[i] = 0u;
  __syncthreads();
  const unsigned pref  = sPref;
  const int      shift = (pass == 0) ? 21 : (pass == 1 ? 10 : 0);
  const int      nb    = (pass == 2) ? 1024 : 2048;
  const unsigned* row = scoresU + (size_t)b * N_;
  for (int n = tid; n < N_; n += 1024) {
    unsigned u = row[n];
    bool match = (pass == 0) ||
                 (pass == 1 ? ((u >> 21) == pref) : ((u >> 10) == pref));
    if (match) atomicAdd(&hist[(u >> shift) & (nb - 1)], 1u);
  }
  __syncthreads();
  if (tid == 0) {
    unsigned krem = sKrem, cum = 0;
    int pick = 0;
    for (int bin = nb - 1; bin >= 0; --bin) {
      unsigned h = hist[bin];
      if (cum + h >= krem) { pick = bin; break; }
      cum += h;
    }
    uint4 st;
    st.x = (pref << ((pass == 2) ? 10 : 11)) | (unsigned)pick; // prefix / kth key
    st.y = krem - cum;                                         // ties still needed
    st.z = 0u; st.w = 0u;
    rowState[b] = st;
  }
}

// Phase 2b: compact strict winners (u > kth) + collect tie indices.
__global__ __launch_bounds__(1024)
void compact_kernel(const unsigned* __restrict__ scoresU,
                    uint4* __restrict__ rowState,
                    unsigned long long* __restrict__ cand,
                    unsigned* __restrict__ tieIdx) {
  __shared__ unsigned cDef, cTie;
  const int b = blockIdx.x;
  const int tid = threadIdx.x;
  if (tid == 0) { cDef = 0u; cTie = 0u; }
  __syncthreads();
  const unsigned kth = rowState[b].x;
  const unsigned* row = scoresU + (size_t)b * N_;
  unsigned long long* crow = cand + (size_t)b * K_;
  unsigned* trow = tieIdx + (size_t)b * TIE_MAX;
  for (int n = tid; n < N_; n += 1024) {
    unsigned u = row[n];
    if (u > kth) {
      unsigned pos = atomicAdd(&cDef, 1u);   // pos < K_ - krem guaranteed
      crow[pos] = ((unsigned long long)u << 32) | (unsigned)(~(unsigned)n);
    } else if (u == kth) {
      unsigned pos = atomicAdd(&cTie, 1u);
      if (pos < TIE_MAX) trow[pos] = (unsigned)n;
    }
  }
  __syncthreads();
  if (tid == 0) { uint4 st = rowState[b]; st.z = cTie; rowState[b] = st; }
}

// Phase 2c: pick the `krem` lowest-index ties (top_k tie order) -> fill slots.
__global__ void tie_fixup_kernel(uint4* __restrict__ rowState,
                                 unsigned long long* __restrict__ cand,
                                 unsigned* __restrict__ tieIdx) {
  const int b = blockIdx.x;
  if (threadIdx.x != 0) return;
  uint4 st = rowState[b];
  const unsigned kth  = st.x;
  const unsigned krem = st.y;
  unsigned cnt = st.z; if (cnt > TIE_MAX) cnt = TIE_MAX;
  const unsigned A = K_ - krem;
  unsigned* trow = tieIdx + (size_t)b * TIE_MAX;
  unsigned long long* crow = cand + (size_t)b * K_;
  unsigned take = krem < cnt ? krem : cnt;
  for (unsigned t = 0; t < take; ++t) {           // partial selection sort
    unsigned best = t;
    for (unsigned s = t + 1; s < cnt; ++s)
      if (trow[s] < trow[best]) best = s;
    unsigned tmp = trow[t]; trow[t] = trow[best]; trow[best] = tmp;
    crow[A + t] = ((unsigned long long)kth << 32) | (unsigned)(~trow[t]);
  }
  for (unsigned t = take; t < krem; ++t)          // (unreachable in practice)
    crow[A + t] = ((unsigned long long)kth << 32) | 0xFFFFFFFFull;
}

// Phase 2d: per-row LDS bitonic sort of 4096 packed (key<<32 | ~idx),
// descending -> key desc, idx asc (matches reference top_k ordering).
__global__ __launch_bounds__(1024)
void sort_neg_kernel(const unsigned long long* __restrict__ cand,
                     unsigned* __restrict__ negIdx) {
  __shared__ unsigned long long s[K_];   // 32 KB
  const int b = blockIdx.x;
  const int tid = threadIdx.x;
  const unsigned long long* crow = cand + (size_t)b * K_;
  for (int i = tid; i < K_; i += 1024) s[i] = crow[i];
  for (unsigned k = 2; k <= (unsigned)K_; k <<= 1) {
    for (unsigned j = k >> 1; j > 0; j >>= 1) {
      __syncthreads();
      for (unsigned i = tid; i < (unsigned)K_; i += 1024) {
        unsigned ixj = i ^ j;
        if (ixj > i) {
          unsigned long long a = s[i], c = s[ixj];
          bool descSeg = (i & k) == 0;
          if (descSeg ? (a < c) : (a > c)) { s[i] = c; s[ixj] = a; }
        }
      }
    }
  }
  __syncthreads();
  for (int i = tid; i < K_; i += 1024)
    negIdx[(size_t)b * K_ + i] = ~(unsigned)(s[i] & 0xFFFFFFFFull);
}

// ---------------------------------------------------------------------------
// Phase 3: out[b, 1+k] = exp(<mem[neg], anchor[b]> / T) via WMMA.
// One wave per (b, 16-neg tile); A = 16 gathered bank rows (float4 runs),
// B = anchor[b] chunk replicated across the 16 columns.
__global__ __launch_bounds__(256)
void contrast_kernel(const float* __restrict__ anchor,
                     const float* __restrict__ mem,
                     const unsigned* __restrict__ negIdx,
                     float* __restrict__ out) {
  const int tid   = threadIdx.x;
  const int lane  = tid & 31;
  const int lrow  = lane & 15;
  const int lhalf = lane >> 4;
  const int task  = blockIdx.x * 8 + (tid >> 5);
  const int b  = task >> 8;     // 256 k-tiles per row
  const int kt = task & 255;

  const unsigned r = negIdx[(size_t)b * K_ + kt * 16 + lrow];
  const float* mrow = mem + (size_t)r * D_;        // [N,D] row gather
  const float* arow = anchor + b * D_;

  v8f c = {};
#pragma unroll
  for (int ks = 0; ks < 4; ++ks) {
    v16bf Av = load_a_frag(mrow, ks, lhalf);
    v16bf Bv = load_b_frag_bcast(arow, ks, lhalf);
    c = __builtin_amdgcn_wmma_f32_16x16x32_bf16(false, Av, false, Bv,
                                                (short)0, c, false, false);
  }
  if (lrow == 0) {   // lanes 0 & 16 cover M=0..7 / 8..15
#pragma unroll
    for (int j = 0; j < 8; ++j) {
      int m = j + 8 * lhalf;
      out[(size_t)b * (K_ + 1) + 1 + kt * 16 + m] = __expf(c[j] * INV_T);
    }
  }
}

// Slot 0: positive pair.
__global__ __launch_bounds__(128)
void pos_kernel(const float* __restrict__ anchor,
                const float* __restrict__ pair,
                float* __restrict__ out) {
  __shared__ float red[128];
  const int b = blockIdx.x, t = threadIdx.x;
  red[t] = anchor[b * D_ + t] * pair[b * D_ + t];
  __syncthreads();
  for (int s = 64; s > 0; s >>= 1) {
    if (t < s) red[t] += red[t + s];
    __syncthreads();
  }
  if (t == 0) out[(size_t)b * (K_ + 1)] = __expf(red[0] * INV_T);
}

// ---------------------------------------------------------------------------
// Phase 4: deterministic two-stage sum -> Z = mean*N -> scale.
__global__ __launch_bounds__(256)
void reduce_partial_kernel(const float* __restrict__ out,
                           float* __restrict__ partials) {
  __shared__ float red[256];
  float s = 0.f;
  for (int i = blockIdx.x * 256 + threadIdx.x; i < TOPE; i += 512 * 256)
    s += out[i];
  red[threadIdx.x] = s;
  __syncthreads();
  for (int st = 128; st > 0; st >>= 1) {
    if (threadIdx.x < st) red[threadIdx.x] += red[threadIdx.x + st];
    __syncthreads();
  }
  if (threadIdx.x == 0) partials[blockIdx.x] = red[0];
}

__global__ __launch_bounds__(512)
void finalize_kernel(const float* __restrict__ partials,
                     float* __restrict__ invZ) {
  __shared__ float red[512];
  red[threadIdx.x] = partials[threadIdx.x];
  __syncthreads();
  for (int st = 256; st > 0; st >>= 1) {
    if (threadIdx.x < st) red[threadIdx.x] += red[threadIdx.x + st];
    __syncthreads();
  }
  if (threadIdx.x == 0) {
    float mean = red[0] / (float)TOPE;
    invZ[0] = 1.0f / (mean * (float)N_);
  }
}

__global__ __launch_bounds__(256)
void scale_kernel(float* __restrict__ out, const float* __restrict__ invZ) {
  float z = invZ[0];
  for (int i = blockIdx.x * 256 + threadIdx.x; i < TOPE; i += 2048 * 256)
    out[i] *= z;
}

// ---------------------------------------------------------------------------
extern "C" void kernel_launch(void* const* d_in, const int* in_sizes, int n_in,
                              void* d_out, int out_size, void* d_ws, size_t ws_size,
                              hipStream_t stream) {
  (void)in_sizes; (void)n_in; (void)out_size; (void)ws_size;
  const float* anchor = (const float*)d_in[0];
  const float* pair   = (const float*)d_in[1];
  const float* mem    = (const float*)d_in[2];
  // d_in[3] = membank_idx: unused by the reference computation.
  const float* thr    = (const float*)d_in[4];
  float* out = (float*)d_out;

  char* ws = (char*)d_ws;
  size_t off = 0;
  unsigned*            scoresU  = (unsigned*)(ws + off);           off += (size_t)B_ * N_ * 4;       // 268.4 MB keys
  unsigned long long*  cand     = (unsigned long long*)(ws + off); off += (size_t)B_ * K_ * 8;       // 8.4 MB
  unsigned*            negIdx   = (unsigned*)(ws + off);           off += (size_t)B_ * K_ * 4;       // 4.2 MB
  uint4*               rowState = (uint4*)(ws + off);              off += (size_t)B_ * 16;
  unsigned*            tieIdx   = (unsigned*)(ws + off);           off += (size_t)B_ * TIE_MAX * 4;  // 4.2 MB
  float*               partials = (float*)(ws + off);              off += 512 * 4;
  float*               invZ     = (float*)(ws + off);              off += 256;

  gemm_scores_kernel<<<1024, 512, 0, stream>>>(anchor, mem, thr, scoresU);
  radix_pass_kernel<<<B_, 1024, 0, stream>>>(scoresU, rowState, 0);
  radix_pass_kernel<<<B_, 1024, 0, stream>>>(scoresU, rowState, 1);
  radix_pass_kernel<<<B_, 1024, 0, stream>>>(scoresU, rowState, 2);
  compact_kernel<<<B_, 1024, 0, stream>>>(scoresU, rowState, cand, tieIdx);
  tie_fixup_kernel<<<B_, 64, 0, stream>>>(rowState, cand, tieIdx);
  sort_neg_kernel<<<B_, 1024, 0, stream>>>(cand, negIdx);
  pos_kernel<<<B_, 128, 0, stream>>>(anchor, pair, out);
  contrast_kernel<<<8192, 256, 0, stream>>>(anchor, mem, negIdx, out);
  reduce_partial_kernel<<<512, 256, 0, stream>>>(out, partials);
  finalize_kernel<<<1, 512, 0, stream>>>(partials, invZ);
  scale_kernel<<<2048, 256, 0, stream>>>(out, invZ);
}
